// FFWRelativeSelfAttentionModule_67731634258345
// MI455X (gfx1250) — matrix operand order repaired and checked by reading, not compiled, not run on hardware
//
#include <hip/hip_runtime.h>
#include <hip/hip_bf16.h>

// ---------------- problem constants ----------------
#define NUM_LAYERS 4
#define NUM_HEADS  12
#define EMBED      768
#define BATCH      4
#define SEQ        1024
#define HEAD_D     64
#define MROWS      (BATCH * SEQ)          // 4096 token rows
#define LN_EPS     1e-5f

typedef unsigned short ushortT;
typedef __attribute__((ext_vector_type(16))) __bf16 v16bf;
typedef __attribute__((ext_vector_type(8)))  float  v8f;
typedef __attribute__((ext_vector_type(4)))  unsigned int u32x4;

union Frag { v16bf v; u32x4 q[2]; };

__device__ __forceinline__ v8f wmma_bf16(v16bf a, v16bf b, v8f c) {
    // D(f32 16x16) = A(bf16 16x32) * B(bf16 32x16) + C
    return __builtin_amdgcn_wmma_f32_16x16x32_bf16(
        false, a, false, b, (short)0, c, false, false);
}

// f32 -> bf16 bits, round-to-nearest-even
__device__ __forceinline__ ushortT f2bf(float f) {
    union { float f; unsigned u; } x; x.f = f;
    unsigned u = x.u;
    u += 0x7FFFu + ((u >> 16) & 1u);
    return (ushortT)(u >> 16);
}

// ---------------- bulk f32 -> bf16 convert ----------------
__global__ void k_cvt(const float* __restrict__ s, ushortT* __restrict__ d, long n) {
    long i   = (long)blockIdx.x * blockDim.x + threadIdx.x;
    long str = (long)gridDim.x * blockDim.x;
    for (; i < n; i += str) d[i] = f2bf(s[i]);
}

// ---------------- tiled bf16 WMMA GEMM ----------------
// C[M,N] = A[M,K](bf16,row-major) * W[N,K]^T(bf16,row-major) + bias (+resid)(+relu)
// Block tile 128x128, K-step 32. 8 waves: 4(M) x 2(N); wave tile 32x64 = 2x4 WMMA tiles
// (8 wmma per K-step per wave against 6 LDS fragment gathers).
#define GBM 128
#define GBN 128
#define GBK 32
#define APITCH 40   // halves: 32 + 8 pad (16B) to stagger banks
#define BPITCH 40

__global__ __launch_bounds__(256) void k_gemm(
    const ushortT* __restrict__ A, const ushortT* __restrict__ W,
    const float* __restrict__ bias, const float* __restrict__ resid,
    float* __restrict__ Cf, ushortT* __restrict__ Cb,
    int M, int N, int K, int relu)
{
    __shared__ __attribute__((aligned(16))) ushortT As[GBM * APITCH];
    __shared__ __attribute__((aligned(16))) ushortT Bs[GBN * BPITCH];

    const int tid  = threadIdx.x;
    const int lane = tid & 31;
    const int wave = tid >> 5;
    const int wm   = wave >> 1;      // 0..3  (rows)
    const int wn   = wave & 1;       // 0..1  (cols, 64 each)
    const int l15  = lane & 15;
    const int hl   = lane >> 4;      // lane-half select
    const int m0   = blockIdx.y * GBM;
    const int n0   = blockIdx.x * GBN;

    const int srow = tid >> 1;               // staging row (0..127)
    const int soff = (tid & 1) * 16;         // staging halves offset (0|16)

    v8f acc[2][4] = {};

    for (int k0 = 0; k0 < K; k0 += GBK) {
        // prefetch next K-tile chunks (global_prefetch_b8 path, L2-resident weights)
        if (k0 + GBK < K) {
            __builtin_prefetch(A + (size_t)(m0 + srow) * K + k0 + GBK + soff, 0, 0);
            __builtin_prefetch(W + (size_t)(n0 + srow) * K + k0 + GBK + soff, 0, 0);
        }
        // stage A tile: 128 rows x 32 halves; 2 threads/row, 16 halves (32B) each
        {
            const u32x4* src = (const u32x4*)(A + (size_t)(m0 + srow) * K + k0 + soff);
            u32x4* dst = (u32x4*)(As + srow * APITCH + soff);
            dst[0] = src[0];
            dst[1] = src[1];
        }
        // stage W tile: 128 rows x 32 halves; same pattern
        {
            const u32x4* src = (const u32x4*)(W + (size_t)(n0 + srow) * K + k0 + soff);
            u32x4* dst = (u32x4*)(Bs + srow * BPITCH + soff);
            dst[0] = src[0];
            dst[1] = src[1];
        }
        __syncthreads();

        // A fragment (ISA 16-bit A 16x32): lane=row; lanes0-15 K{0-7,16-23}, lanes16-31 K{8-15,24-31}
        Frag a[2], b[4];
        for (int i = 0; i < 2; ++i) {
            const ushortT* p = As + (wm * 32 + i * 16 + l15) * APITCH + hl * 8;
            a[i].q[0] = *(const u32x4*)(p);
            a[i].q[1] = *(const u32x4*)(p + 16);
        }
        // B fragment (32x16): lane=column; lanes0-15 K{0-7,16-23}, lanes16-31 K{8-15,24-31}
        for (int j = 0; j < 4; ++j) {
            const ushortT* p = Bs + (wn * 64 + j * 16 + l15) * BPITCH + hl * 8;
            b[j].q[0] = *(const u32x4*)(p);
            b[j].q[1] = *(const u32x4*)(p + 16);
        }
        for (int i = 0; i < 2; ++i)
            for (int j = 0; j < 4; ++j)
                acc[i][j] = wmma_bf16(a[i].v, b[j].v, acc[i][j]);
        __syncthreads();
    }

    // epilogue: C/D layout VGPR r -> (row r | r+8), col = lane&15
    for (int i = 0; i < 2; ++i) {
        for (int j = 0; j < 4; ++j) {
            int gm  = m0 + wm * 32 + i * 16;
            int col = n0 + wn * 64 + j * 16 + l15;
            float bv = bias ? bias[col] : 0.0f;
            for (int r = 0; r < 8; ++r) {
                int row = gm + r + hl * 8;
                float v = acc[i][j][r] + bv;
                if (resid) v += resid[(size_t)row * N + col];
                if (relu)  v = fmaxf(v, 0.0f);
                if (Cf) Cf[(size_t)row * N + col] = v;
                if (Cb) Cb[(size_t)row * N + col] = f2bf(v);
            }
        }
    }
}

// ---------------- rotary + scale + layout + bf16 ----------------
// qkv: [B,N,3E] f32.  q,k -> [B,H,N,d] bf16 (rotary applied, q scaled).
// v -> [B,H,d,N] bf16 (transposed so PV B-fragments are LDS-row contiguous).
__global__ __launch_bounds__(384) void k_qkv_post(
    const float* __restrict__ qkv, const float* __restrict__ pe,
    ushortT* __restrict__ q_r, ushortT* __restrict__ k_r, ushortT* __restrict__ v_t)
{
    const int bn = blockIdx.x;            // b*SEQ + n
    const int n  = bn & (SEQ - 1);
    const int b  = bn >> 10;
    const int t  = threadIdx.x;           // pair index, 0..383
    const int e0 = 2 * t, e1 = e0 + 1;

    const float* base = qkv + (size_t)bn * (3 * EMBED);
    float q0 = base[e0],             q1 = base[e1];
    float k0 = base[EMBED + e0],     k1 = base[EMBED + e1];
    float v0 = base[2 * EMBED + e0], v1 = base[2 * EMBED + e1];

    const float* pep = pe + ((size_t)bn * EMBED + e0) * 2;
    float c0 = pep[0], s0 = pep[1], c1 = pep[2], s1 = pep[3];

    const float sc = 0.125f;              // d^-0.5, d=64
    q0 *= sc; q1 *= sc;
    // out[2i] = x[2i]*cos - x[2i+1]*sin ; out[2i+1] = x[2i+1]*cos + x[2i]*sin
    float qr0 = q0 * c0 - q1 * s0;
    float qr1 = q1 * c1 + q0 * s1;
    float kr0 = k0 * c0 - k1 * s0;
    float kr1 = k1 * c1 + k0 * s1;

    const int h  = e0 >> 6;
    const int d0 = e0 & 63;
    size_t qo = (((size_t)b * NUM_HEADS + h) * SEQ + n) * HEAD_D + d0;
    q_r[qo]     = f2bf(qr0);
    q_r[qo + 1] = f2bf(qr1);
    k_r[qo]     = f2bf(kr0);
    k_r[qo + 1] = f2bf(kr1);
    size_t vo = (((size_t)b * NUM_HEADS + h) * HEAD_D + d0) * SEQ + n;
    v_t[vo]       = f2bf(v0);
    v_t[vo + SEQ] = f2bf(v1);
}

// ---------------- flash attention (bf16 WMMA, f32 online softmax) ----------------
// Grid: B*H*(SEQ/64). Block: 128 threads (4 waves); wave owns 16 q-rows.
__global__ __launch_bounds__(128) void k_attn(
    const ushortT* __restrict__ q_r, const ushortT* __restrict__ k_r,
    const ushortT* __restrict__ v_t, ushortT* __restrict__ attn_b)
{
    __shared__ __attribute__((aligned(16))) ushortT Ks[64 * 64];       // [key][d]
    __shared__ __attribute__((aligned(16))) ushortT Vt[64 * 64];       // [d][key]
    __shared__ __attribute__((aligned(16))) ushortT Ps[4][16 * 64];    // per-wave P tile

    const int tid  = threadIdx.x;
    const int lane = tid & 31;
    const int wave = tid >> 5;
    const int l15  = lane & 15;
    const int hl   = lane >> 4;

    const int qt = blockIdx.x & 15;
    const int bh = blockIdx.x >> 4;            // b*H + h
    const int h  = bh % NUM_HEADS;
    const int b  = bh / NUM_HEADS;

    const ushortT* qbase = q_r + (size_t)bh * SEQ * HEAD_D;
    const ushortT* kbase = k_r + (size_t)bh * SEQ * HEAD_D;
    const ushortT* vbase = v_t + (size_t)bh * HEAD_D * SEQ;

    const int q0 = qt * 64 + wave * 16;

    // Q fragments (reused across all key tiles): 2 K-steps over d=64
    Frag qa[2];
    {
        const ushortT* qrow = qbase + (size_t)(q0 + l15) * HEAD_D;
        for (int s = 0; s < 2; ++s) {
            qa[s].q[0] = *(const u32x4*)(qrow + s * 32 + hl * 8);
            qa[s].q[1] = *(const u32x4*)(qrow + s * 32 + hl * 8 + 16);
        }
    }

    float mrow[8], lrow[8];
    for (int r = 0; r < 8; ++r) { mrow[r] = -1e30f; lrow[r] = 0.0f; }
    v8f o[4] = {};

    for (int kt = 0; kt < SEQ / 64; ++kt) {
        __syncthreads();   // previous iteration done reading tiles
        // stage K tile [key][d] and V^T tile [d][key]: 512 16B chunks each
        for (int idx = tid; idx < 512; idx += 128) {
            int row = idx >> 3, ch = (idx & 7) * 8;
            *(u32x4*)(Ks + row * 64 + ch) =
                *(const u32x4*)(kbase + (size_t)(kt * 64 + row) * HEAD_D + ch);
            *(u32x4*)(Vt + row * 64 + ch) =
                *(const u32x4*)(vbase + (size_t)row * SEQ + kt * 64 + ch);
        }
        __syncthreads();

        // scores: S(16x64) = Q(16x64) * K^T ; 4 key sub-tiles, 2 chained K-steps
        v8f s[4] = {};
        for (int ss = 0; ss < 2; ++ss) {
            for (int j = 0; j < 4; ++j) {
                Frag kb;
                const ushortT* p = Ks + (j * 16 + l15) * 64 + ss * 32 + hl * 8;
                kb.q[0] = *(const u32x4*)(p);
                kb.q[1] = *(const u32x4*)(p + 16);
                s[j] = wmma_bf16(qa[ss].v, kb.v, s[j]);
            }
        }

        // online softmax: C/D layout -> lanes0-15 carry rows r, lanes16-31 rows r+8;
        // row reductions are shuffles over 16-lane halves (xor 1,2,4,8).
        for (int r = 0; r < 8; ++r) {
            float mx = s[0][r];
            mx = fmaxf(mx, s[1][r]); mx = fmaxf(mx, s[2][r]); mx = fmaxf(mx, s[3][r]);
            mx = fmaxf(mx, __shfl_xor(mx, 1, 32));
            mx = fmaxf(mx, __shfl_xor(mx, 2, 32));
            mx = fmaxf(mx, __shfl_xor(mx, 4, 32));
            mx = fmaxf(mx, __shfl_xor(mx, 8, 32));
            float mn  = fmaxf(mrow[r], mx);
            float scl = __expf(mrow[r] - mn);
            mrow[r] = mn;
            float rs = 0.0f;
            for (int j = 0; j < 4; ++j) {
                float p = __expf(s[j][r] - mn);
                s[j][r] = p;
                rs += p;
            }
            rs += __shfl_xor(rs, 1, 32);
            rs += __shfl_xor(rs, 2, 32);
            rs += __shfl_xor(rs, 4, 32);
            rs += __shfl_xor(rs, 8, 32);
            lrow[r] = lrow[r] * scl + rs;
            for (int j = 0; j < 4; ++j) o[j][r] *= scl;
        }

        // scatter P (bf16) to per-wave LDS in row-major [qrow][key]
        ushortT* P = Ps[wave];
        for (int j = 0; j < 4; ++j)
            for (int r = 0; r < 8; ++r)
                P[(r + hl * 8) * 64 + j * 16 + l15] = f2bf(s[j][r]);

        // O += P(16x64) * V(64x64): gather P as A-fragments, V^T rows as B-fragments
        for (int ss = 0; ss < 2; ++ss) {
            Frag pa;
            const ushortT* pp = P + l15 * 64 + ss * 32 + hl * 8;
            pa.q[0] = *(const u32x4*)(pp);
            pa.q[1] = *(const u32x4*)(pp + 16);
            for (int j = 0; j < 4; ++j) {
                Frag vb;
                const ushortT* vp = Vt + (j * 16 + l15) * 64 + ss * 32 + hl * 8;
                vb.q[0] = *(const u32x4*)(vp);
                vb.q[1] = *(const u32x4*)(vp + 16);
                o[j] = wmma_bf16(pa.v, vb.v, o[j]);
            }
        }
    }

    // normalize by softmax denominator, emit bf16 [B,N,E] for the Wo GEMM
    for (int j = 0; j < 4; ++j) {
        for (int r = 0; r < 8; ++r) {
            int qrow = q0 + r + hl * 8;
            int col  = h * HEAD_D + j * 16 + l15;
            float v  = o[j][r] / lrow[r];
            attn_b[((size_t)b * SEQ + qrow) * EMBED + col] = f2bf(v);
        }
    }
}

// ---------------- layernorm (row = token), emits f32 (+optional copy) and bf16 ----------------
__global__ __launch_bounds__(256) void k_layernorm(
    const float* __restrict__ x, const float* __restrict__ g, const float* __restrict__ bb,
    float* __restrict__ outf, ushortT* __restrict__ outb, float* __restrict__ outf2)
{
    __shared__ float sbuf[16];
    const int row = blockIdx.x;
    const int tid = threadIdx.x;
    const float* xr = x + (size_t)row * EMBED;

    float v0 = xr[tid], v1 = xr[tid + 256], v2 = xr[tid + 512];
    float s  = v0 + v1 + v2;
    float sq = v0 * v0 + v1 * v1 + v2 * v2;
    for (int m = 1; m < 32; m <<= 1) {
        s  += __shfl_xor(s,  m, 32);
        sq += __shfl_xor(sq, m, 32);
    }
    const int lane = tid & 31, wv = tid >> 5;
    if (lane == 0) { sbuf[wv] = s; sbuf[8 + wv] = sq; }
    __syncthreads();
    if (tid == 0) {
        float S = 0.0f, Q = 0.0f;
        for (int i = 0; i < 8; ++i) { S += sbuf[i]; Q += sbuf[8 + i]; }
        sbuf[0] = S; sbuf[8] = Q;
    }
    __syncthreads();
    float mean = sbuf[0] * (1.0f / EMBED);
    float var  = sbuf[8] * (1.0f / EMBED) - mean * mean;
    float inv  = rsqrtf(var + LN_EPS);

    float vv[3] = { v0, v1, v2 };
    for (int c = 0; c < 3; ++c) {
        int e = tid + c * 256;
        float y = (vv[c] - mean) * inv * g[e] + bb[e];
        outf[(size_t)row * EMBED + e] = y;
        outb[(size_t)row * EMBED + e] = f2bf(y);
        if (outf2) outf2[(size_t)row * EMBED + e] = y;
    }
}

// ---------------- host orchestration ----------------
extern "C" void kernel_launch(void* const* d_in, const int* in_sizes, int n_in,
                              void* d_out, int out_size, void* d_ws, size_t ws_size,
                              hipStream_t stream) {
    const float* query = (const float*)d_in[0];
    const float* qpos  = (const float*)d_in[1];
    const float* Wqkv  = (const float*)d_in[2];
    const float* bqkv  = (const float*)d_in[3];
    const float* Wo    = (const float*)d_in[4];
    const float* bo    = (const float*)d_in[5];
    const float* ln1g  = (const float*)d_in[6];
    const float* ln1b  = (const float*)d_in[7];
    const float* W1    = (const float*)d_in[8];
    const float* b1    = (const float*)d_in[9];
    const float* W2    = (const float*)d_in[10];
    const float* b2    = (const float*)d_in[11];
    const float* ln2g  = (const float*)d_in[12];
    const float* ln2b  = (const float*)d_in[13];
    float* out = (float*)d_out;

    // workspace arena
    char* ws = (char*)d_ws;
    size_t off = 0;
    auto alloc = [&](size_t bytes) -> char* {
        char* p = ws + off;
        off = (off + bytes + 255) & ~(size_t)255;
        return p;
    };
    const size_t nWqkv = (size_t)NUM_LAYERS * 3 * EMBED * EMBED;
    const size_t nWsq  = (size_t)NUM_LAYERS * EMBED * EMBED;
    const size_t nTok  = (size_t)MROWS * EMBED;

    ushortT* wqkv_b = (ushortT*)alloc(nWqkv * 2);
    ushortT* wo_b   = (ushortT*)alloc(nWsq * 2);
    ushortT* w1_b   = (ushortT*)alloc(nWsq * 2);
    ushortT* w2_b   = (ushortT*)alloc(nWsq * 2);
    float*   xbuf   = (float*)alloc(nTok * 4);            // residual stream (f32)
    ushortT* xb     = (ushortT*)alloc(nTok * 2);          // residual stream (bf16)
    float*   qkv    = (float*)alloc((size_t)MROWS * 3 * EMBED * 4);
    ushortT* q_r    = (ushortT*)alloc(nTok * 2);          // [B,H,N,d]
    ushortT* k_r    = (ushortT*)alloc(nTok * 2);          // [B,H,N,d]
    ushortT* v_t    = (ushortT*)alloc(nTok * 2);          // [B,H,d,N]
    ushortT* attn_b = (ushortT*)alloc(nTok * 2);          // attention out (bf16)
    float*   ybuf   = (float*)alloc(nTok * 4);            // pre-LN sum (f32)
    ushortT* h_b    = (ushortT*)alloc(nTok * 2);          // FFN hidden (bf16)

    // one-time weight conversion + residual-stream init
    k_cvt<<<2048, 256, 0, stream>>>(Wqkv, wqkv_b, (long)nWqkv);
    k_cvt<<<2048, 256, 0, stream>>>(Wo,   wo_b,   (long)nWsq);
    k_cvt<<<2048, 256, 0, stream>>>(W1,   w1_b,   (long)nWsq);
    k_cvt<<<2048, 256, 0, stream>>>(W2,   w2_b,   (long)nWsq);
    hipMemcpyAsync(xbuf, query, nTok * 4, hipMemcpyDeviceToDevice, stream);
    k_cvt<<<2048, 256, 0, stream>>>(query, xb, (long)nTok);

    const dim3 gQKV(3 * EMBED / GBN, MROWS / GBM);   // 18 x 32
    const dim3 gSQ(EMBED / GBN, MROWS / GBM);        // 6 x 32

    for (int i = 0; i < NUM_LAYERS; ++i) {
        // QKV projection
        k_gemm<<<gQKV, 256, 0, stream>>>(xb, wqkv_b + (size_t)i * 3 * EMBED * EMBED,
                                         bqkv + (size_t)i * 3 * EMBED, nullptr,
                                         qkv, nullptr, MROWS, 3 * EMBED, EMBED, 0);
        // rotary + scale + head layout + bf16
        k_qkv_post<<<BATCH * SEQ, 384, 0, stream>>>(qkv, qpos, q_r, k_r, v_t);
        // flash attention
        k_attn<<<BATCH * NUM_HEADS * (SEQ / 64), 128, 0, stream>>>(q_r, k_r, v_t, attn_b);
        // output projection + residual
        k_gemm<<<gSQ, 256, 0, stream>>>(attn_b, wo_b + (size_t)i * EMBED * EMBED,
                                        bo + (size_t)i * EMBED, xbuf,
                                        ybuf, nullptr, MROWS, EMBED, EMBED, 0);
        // LN1 -> new residual stream (f32 + bf16)
        k_layernorm<<<MROWS, 256, 0, stream>>>(ybuf, ln1g + (size_t)i * EMBED,
                                               ln1b + (size_t)i * EMBED, xbuf, xb, nullptr);
        // FFN: relu(x W1^T + b1) -> bf16 hidden
        k_gemm<<<gSQ, 256, 0, stream>>>(xb, w1_b + (size_t)i * EMBED * EMBED,
                                        b1 + (size_t)i * EMBED, nullptr,
                                        nullptr, h_b, MROWS, EMBED, EMBED, 1);
        // FFN: h W2^T + b2 + residual
        k_gemm<<<gSQ, 256, 0, stream>>>(h_b, w2_b + (size_t)i * EMBED * EMBED,
                                        b2 + (size_t)i * EMBED, xbuf,
                                        ybuf, nullptr, MROWS, EMBED, EMBED, 0);
        // LN2 -> residual stream + this layer's output slice
        k_layernorm<<<MROWS, 256, 0, stream>>>(ybuf, ln2g + (size_t)i * EMBED,
                                               ln2b + (size_t)i * EMBED, xbuf, xb,
                                               out + (size_t)i * nTok);
    }
    (void)in_sizes; (void)n_in; (void)out_size; (void)ws_size;
}
